// LSTMBranch_89859305767540
// MI455X (gfx1250) — matrix operand order repaired and checked by reading, not compiled, not run on hardware
//
#include <hip/hip_runtime.h>

typedef __bf16 bf16_t;
typedef __attribute__((ext_vector_type(16))) __bf16 v16bf;
typedef __attribute__((ext_vector_type(8)))  __bf16 v8bf;
typedef __attribute__((ext_vector_type(8)))  float   v8f;

#define BATCH   32
#define TSTEPS  2048
#define FEAT    512
#define HID     512
#define H4      2048
#define NCLS    64
#define NBLK    16          // each block owns 32 hidden columns (16*32 = 512)
#define NTHR    256         // 8 waves: wave = gate (0..3) + 4*jtile (0..1)
#define LDA     (512 + 8)   // bf16 halves per staged row (pad vs 64-bank LDS)

__device__ __forceinline__ bf16_t f2bf(float f) {
    unsigned u = __float_as_uint(f);
    u += 0x7FFFu + ((u >> 16) & 1u);          // round-to-nearest-even
    unsigned short s = (unsigned short)(u >> 16);
    return __builtin_bit_cast(bf16_t, s);
}

__device__ __forceinline__ float bf2f(bf16_t v) {
    unsigned u = (unsigned)__builtin_bit_cast(unsigned short, v) << 16;
    return __uint_as_float(u);
}

__device__ __forceinline__ float sigmoidf(float x) {
    return 1.0f / (1.0f + __expf(-x));
}

// ---------------------------------------------------------------------------
// zero bf16 h state (h0a,h0b,h1a,h1b) and the grid-barrier counter
// ---------------------------------------------------------------------------
__global__ void init_state_kernel(unsigned* hwords, unsigned* ctr) {
    int i = blockIdx.x * blockDim.x + threadIdx.x;
    if (i < 4 * BATCH * HID / 2) hwords[i] = 0u;     // 4 buffers of bf16 zeros
    if (i == 0) *ctr = 0u;
}

// ---------------------------------------------------------------------------
// fp32 [K=512][N=2048] -> bf16 [N=2048][K=512]  (transposed for WMMA B-frags)
// blockIdx.y selects which of the 4 weight matrices
// ---------------------------------------------------------------------------
__global__ void convert_transpose_kernel(const float* __restrict__ W0,
                                         const float* __restrict__ W1,
                                         const float* __restrict__ W2,
                                         const float* __restrict__ W3,
                                         bf16_t* __restrict__ D0,
                                         bf16_t* __restrict__ D1,
                                         bf16_t* __restrict__ D2,
                                         bf16_t* __restrict__ D3) {
    const float* S;
    bf16_t* D;
    switch (blockIdx.y) {
        case 0:  S = W0; D = D0; break;
        case 1:  S = W1; D = D1; break;
        case 2:  S = W2; D = D2; break;
        default: S = W3; D = D3; break;
    }
    int e = blockIdx.x * blockDim.x + threadIdx.x;       // over 512*2048
    if (e < 512 * 2048) {
        int k = e >> 11;            // row in source (K)
        int n = e & 2047;           // col in source (N)
        D[(size_t)n * 512 + k] = f2bf(S[e]);
    }
}

// ---------------------------------------------------------------------------
// persistent LSTM kernel: 16 blocks loop over all 2048 timesteps, 2 layers
// ---------------------------------------------------------------------------
__global__ void __launch_bounds__(NTHR, 1)
lstm_persistent_kernel(const float* __restrict__ feat,
                       const bf16_t* __restrict__ Wi0t, const bf16_t* __restrict__ Wh0t,
                       const float* __restrict__ bh0,
                       const bf16_t* __restrict__ Wi1t, const bf16_t* __restrict__ Wh1t,
                       const float* __restrict__ bh1,
                       bf16_t* __restrict__ h0a, bf16_t* __restrict__ h0b,
                       bf16_t* __restrict__ h1a, bf16_t* __restrict__ h1b,
                       unsigned* __restrict__ ctr) {
    __shared__ bf16_t sA[2][BATCH][LDA];    // dual-buffered staged activations
    __shared__ float  sZ[8][BATCH][16];     // per-wave gate pre-activations

    const int lane  = threadIdx.x & 31;
    const int wave  = threadIdx.x >> 5;
    const int khalf = lane >> 4;            // lane-half selects contiguous K half (k = 16*khalf + s)
    const int mrow  = lane & 15;            // A row within 16-row fragment
    const int jl    = lane & 15;            // output column within tile
    const int gate  = wave & 3;             // i,f,g,o
    const int jt    = wave >> 2;            // which 16-col tile of the block's 32
    const int jbase = blockIdx.x * 32 + jt * 16;
    const int ncol  = gate * HID + jbase + jl;   // column in z[.,0..2047]

    // c state lives in registers: fixed thread<->(b,j) mapping across all steps
    float cr0[4] = {0.f, 0.f, 0.f, 0.f};
    float cr1[4] = {0.f, 0.f, 0.f, 0.f};

    // async DMA: bf16 [32 x 512] global -> LDS buffer (16B chunks, ASYNCcnt)
    auto stage_async = [&](const bf16_t* __restrict__ src, int buf) {
        for (int i = threadIdx.x; i < BATCH * (HID / 8); i += NTHR) {
            int r  = i >> 6;                // row 0..31
            int c8 = i & 63;                // 16-byte chunk 0..63
            unsigned ldsoff = (unsigned)(uintptr_t)(void*)&sA[buf][r][c8 * 8];
            const void* g = (const void*)(src + (size_t)r * HID + (size_t)c8 * 8);
            asm volatile("global_load_async_to_lds_b128 %0, %1, off"
                         :: "v"(ldsoff), "v"(g) : "memory");
        }
    };

    // fp32 [32 x 512] global -> bf16 LDS buffer 0 (VALU convert)
    auto stage_cvt = [&](const float* __restrict__ src, size_t rstride) {
        for (int i = threadIdx.x; i < BATCH * (HID / 4); i += NTHR) {
            int r  = i >> 7;                // row 0..31
            int c4 = i & 127;               // float4 index 0..127
            const float4 v = *(const float4*)(src + (size_t)r * rstride + (size_t)c4 * 4);
            bf16_t* dst = &sA[0][r][c4 * 4];
            dst[0] = f2bf(v.x); dst[1] = f2bf(v.y);
            dst[2] = f2bf(v.z); dst[3] = f2bf(v.w);
        }
    };

    v8f acc0, acc1;
    // accumulate sA[buf][32x512] @ Wt-column-slice: acc0 rows 0-15, acc1 rows 16-31
    auto gemm = [&](const bf16_t* __restrict__ Wt, int buf) {
#pragma unroll 4
        for (int kc = 0; kc < 16; ++kc) {
            const int k0 = kc * 32;
            const bf16_t* ar0 = &sA[buf][mrow][k0 + khalf * 16];
            const bf16_t* ar1 = &sA[buf][mrow + 16][k0 + khalf * 16];
            v8bf a0l = *(const v8bf*)ar0;
            v8bf a0h = *(const v8bf*)(ar0 + 8);
            v8bf a1l = *(const v8bf*)ar1;
            v8bf a1h = *(const v8bf*)(ar1 + 8);
            const bf16_t* bp = Wt + (size_t)ncol * 512 + k0 + khalf * 16;
            v8bf bl = *(const v8bf*)bp;
            v8bf bh = *(const v8bf*)(bp + 8);
            v16bf a0 = __builtin_shufflevector(a0l, a0h, 0,1,2,3,4,5,6,7,8,9,10,11,12,13,14,15);
            v16bf a1 = __builtin_shufflevector(a1l, a1h, 0,1,2,3,4,5,6,7,8,9,10,11,12,13,14,15);
            v16bf bb = __builtin_shufflevector(bl,  bh,  0,1,2,3,4,5,6,7,8,9,10,11,12,13,14,15);
            acc0 = __builtin_amdgcn_wmma_f32_16x16x32_bf16(false, a0, false, bb,
                                                           (short)0, acc0, false, false);
            acc1 = __builtin_amdgcn_wmma_f32_16x16x32_bf16(false, a1, false, bb,
                                                           (short)0, acc1, false, false);
        }
    };

    unsigned barTarget = 0;

    for (int t = 0; t < TSTEPS; ++t) {
        bf16_t* h0cur = (t & 1) ? h0b : h0a;
        bf16_t* h0nxt = (t & 1) ? h0a : h0b;
        bf16_t* h1cur = (t & 1) ? h1b : h1a;
        bf16_t* h1nxt = (t & 1) ? h1a : h1b;

        for (int layer = 0; layer < 2; ++layer) {
            const bf16_t* Wa   = layer ? Wi1t : Wi0t;
            const bf16_t* Wb   = layer ? Wh1t : Wh0t;
            const float*  bias = layer ? bh1 : bh0;
            bf16_t*       hnx  = layer ? h1nxt : h0nxt;
            float*        creg = layer ? cr1 : cr0;

            const float bv = bias[ncol];
            acc0 = (v8f){bv, bv, bv, bv, bv, bv, bv, bv};
            acc1 = acc0;

            // stage both GEMM inputs; async h-copy overlaps feature conversion
            stage_async(layer ? h1cur : h0cur, 1);          // XB (recurrent h)
            if (layer == 0)
                stage_cvt(feat + (size_t)t * FEAT, (size_t)TSTEPS * FEAT);
            else
                stage_async(h0nxt, 0);                      // layer1 input = h0_new
            asm volatile("s_wait_asynccnt 0x0" ::: "memory");
            __syncthreads();

            gemm(Wa, 0);
            gemm(Wb, 1);

            // spill gate pre-activations: D layout M = v + 8*khalf (+16 for acc1)
#pragma unroll
            for (int v = 0; v < 8; ++v) {
                sZ[wave][khalf * 8 + v][jl]      = acc0[v];
                sZ[wave][16 + khalf * 8 + v][jl] = acc1[v];
            }
            __syncthreads();

            // fused elementwise LSTM update; c stays in registers, h -> bf16
#pragma unroll
            for (int q = 0; q < 4; ++q) {
                int e   = threadIdx.x + q * NTHR;
                int jt2 = e >> 9;
                int rem = e & 511;
                int b   = rem >> 4;
                int j   = rem & 15;
                int wb  = jt2 * 4;
                float zi = sZ[wb + 0][b][j];
                float zf = sZ[wb + 1][b][j];
                float zg = sZ[wb + 2][b][j];
                float zo = sZ[wb + 3][b][j];
                int col = blockIdx.x * 32 + jt2 * 16 + j;
                float cn = sigmoidf(zf) * creg[q] + sigmoidf(zi) * tanhf(zg);
                creg[q] = cn;
                hnx[b * HID + col] = f2bf(sigmoidf(zo) * tanhf(cn));
            }

            // ---- grid-wide barrier (16 co-resident blocks, atomic generation) ----
            barTarget += NBLK;
            __threadfence();
            __syncthreads();
            if (threadIdx.x == 0) {
                atomicAdd(ctr, 1u);
                while (*(volatile unsigned*)ctr < barTarget)
                    __builtin_amdgcn_s_sleep(1);
            }
            __syncthreads();
            __threadfence();
        }
    }
}

// ---------------------------------------------------------------------------
// output head: out[32,64] = h1(bf16) @ Wout + bout
// ---------------------------------------------------------------------------
__global__ void head_kernel(const bf16_t* __restrict__ h1,
                            const float* __restrict__ Wout,
                            const float* __restrict__ bout,
                            float* __restrict__ out) {
    int i = blockIdx.x * blockDim.x + threadIdx.x;
    if (i >= BATCH * NCLS) return;
    int b = i >> 6;
    int n = i & 63;
    float acc = bout[n];
#pragma unroll 8
    for (int k = 0; k < HID; ++k)
        acc += bf2f(h1[b * HID + k]) * Wout[k * NCLS + n];
    out[i] = acc;
}

// ---------------------------------------------------------------------------
extern "C" void kernel_launch(void* const* d_in, const int* in_sizes, int n_in,
                              void* d_out, int out_size, void* d_ws, size_t ws_size,
                              hipStream_t stream) {
    const float* feat = (const float*)d_in[0];
    const float* Wi0  = (const float*)d_in[1];
    const float* Wh0  = (const float*)d_in[2];
    const float* bh0  = (const float*)d_in[3];
    const float* Wi1  = (const float*)d_in[4];
    const float* Wh1  = (const float*)d_in[5];
    const float* bh1  = (const float*)d_in[6];
    const float* Wout = (const float*)d_in[7];
    const float* bout = (const float*)d_in[8];
    float* out = (float*)d_out;

    // workspace: 4 transposed bf16 weights (2MB each), 4 bf16 h buffers
    // (32KB each), one barrier counter.
    const size_t WELT = (size_t)512 * 2048;       // elements per weight matrix
    bf16_t* Wi0t = (bf16_t*)d_ws;
    bf16_t* Wh0t = Wi0t + WELT;
    bf16_t* Wi1t = Wh0t + WELT;
    bf16_t* Wh1t = Wi1t + WELT;
    bf16_t* h0a  = Wh1t + WELT;
    bf16_t* h0b  = h0a + BATCH * HID;
    bf16_t* h1a  = h0b + BATCH * HID;
    bf16_t* h1b  = h1a + BATCH * HID;
    unsigned* ctr = (unsigned*)(h1b + BATCH * HID);

    init_state_kernel<<<(4 * BATCH * HID / 2 + 255) / 256, 256, 0, stream>>>(
        (unsigned*)h0a, ctr);

    convert_transpose_kernel<<<dim3((512 * 2048 + 255) / 256, 4), 256, 0, stream>>>(
        Wi0, Wh0, Wi1, Wh1, Wi0t, Wh0t, Wi1t, Wh1t);

    lstm_persistent_kernel<<<NBLK, NTHR, 0, stream>>>(
        feat, Wi0t, Wh0t, bh0, Wi1t, Wh1t, bh1,
        h0a, h0b, h1a, h1b, ctr);

    // t = 2047 is odd -> final h1 lives in h1a
    head_kernel<<<(BATCH * NCLS + 255) / 256, 256, 0, stream>>>(h1a, Wout, bout, out);
}
